// GQA_58884001628578
// MI455X (gfx1250) — matrix operand order repaired and checked by reading, not compiled, not run on hardware
//
#include <hip/hip_runtime.h>
#include <hip/hip_bf16.h>
#include <hip/hip_fp16.h>

// GQA attention block for MI455X (gfx1250, wave32, WMMA).
// v3: attention blocks = 256 threads / 8 waves / 128 query rows with
// register double-buffered K/V staging and hoisted Q fragments; wave-uniform
// causal early-out; GEMMs keep 128x128x64 tiles with reg double-buffering.

typedef __attribute__((ext_vector_type(16))) _Float16 v16h;
typedef __attribute__((ext_vector_type(8)))  _Float16 v8h;
typedef __attribute__((ext_vector_type(4)))  _Float16 v4h;
typedef __attribute__((ext_vector_type(8)))  float    v8f;

#define WMMA_F16(a, b, c) \
  __builtin_amdgcn_wmma_f32_16x16x32_f16(false, (a), false, (b), (short)0, (c), false, false)

// ---------------------------------------------------------------------------
// Fragment loaders (LDS -> VGPR) matching the CDNA5 16-bit WMMA layouts.
// A (16x32, row-major [row][k], ld in halves):
//   lanes 0-15  : row=lane,    elems 0-7 -> K=k0+0..7,  elems 8-15 -> K=k0+16..23
//   lanes 16-31 : row=lane-16, elems 0-7 -> K=k0+8..15, elems 8-15 -> K=k0+24..31
static __device__ __forceinline__
v16h frag_a_rm(const _Float16* p, int ld, int row, int k0, int lane) {
  const int lo = lane & 15, hi = lane >> 4;
  const _Float16* base = p + (size_t)(row + lo) * ld + k0 + hi * 8;
  v8h c0 = *(const v8h*)base;
  v8h c1 = *(const v8h*)(base + 16);
  v16h f;
#pragma unroll
  for (int i = 0; i < 8; ++i) { f[i] = c0[i]; f[i + 8] = c1[i]; }
  return f;
}

// B (32x16) from an [N][K] (n-major) LDS buffer:
//   lanes 0-15 : N=n0+lane, K=k0+0..7,16..23 ; lanes 16-31 : N=n0+lane-16, K=k0+8..15,24..31
static __device__ __forceinline__
v16h frag_b_nm(const _Float16* p, int ld, int n0, int k0, int lane) {
  const int lo = lane & 15, hi = lane >> 4;
  const _Float16* base = p + (size_t)(n0 + lo) * ld + k0 + hi * 8;
  v8h c0 = *(const v8h*)base;
  v8h c1 = *(const v8h*)(base + 16);
  v16h f;
#pragma unroll
  for (int i = 0; i < 8; ++i) { f[i] = c0[i]; f[i + 8] = c1[i]; }
  return f;
}

// ---------------------------------------------------------------------------
__global__ void __launch_bounds__(256)
f32_to_f16_kernel(const float* __restrict__ in, _Float16* __restrict__ out, size_t n) {
  size_t n4 = n >> 2;
  size_t stride = (size_t)gridDim.x * blockDim.x;
  for (size_t i = (size_t)blockIdx.x * blockDim.x + threadIdx.x; i < n4; i += stride) {
    float4 v = ((const float4*)in)[i];
    v4h h;
    h[0] = (_Float16)v.x; h[1] = (_Float16)v.y;
    h[2] = (_Float16)v.z; h[3] = (_Float16)v.w;
    ((v4h*)out)[i] = h;
  }
}

// in: f32 [K][N] row-major -> out: f16 [N][K] row-major (one-time weight prep).
__global__ void __launch_bounds__(256)
transpose_cvt_kernel(const float* __restrict__ in, _Float16* __restrict__ out,
                     int K, int N) {
  constexpr int LDT = 72;                 // 144B row pitch (16B aligned)
  __shared__ _Float16 t[64 * LDT];
  const int k0 = blockIdx.x * 64;
  const int n0 = blockIdx.y * 64;
  const int tid = threadIdx.x;
#pragma unroll
  for (int i = 0; i < 16; ++i) {
    int idx = tid + i * 256;
    int r = idx >> 6;                     // k within tile
    int c = idx & 63;                     // n within tile
    t[c * LDT + r] = (_Float16)in[(size_t)(k0 + r) * N + n0 + c];
  }
  __syncthreads();
#pragma unroll
  for (int i = 0; i < 2; ++i) {
    int idx = (tid + i * 256) * 8;
    int n = idx >> 6;
    int k = idx & 63;
    *(v8h*)&out[(size_t)(n0 + n) * K + k0 + k] = *(const v8h*)&t[n * LDT + k];
  }
}

// ---------------------------------------------------------------------------
// C[M,N] = A[M,K](f16 rm) * Wt[N,K](f16 rm, pre-transposed) + bias(f32).
// 128x128x64 tiles, 8 waves (2 M x 4 N), 16 WMMA per K-step per wave,
// register double-buffering of the next global tile.
template <bool OUT_F32, bool TRANS_OUT>
__global__ void __launch_bounds__(256)
gemm_bias_kernel(const _Float16* __restrict__ A, const _Float16* __restrict__ Wt,
                 const float* __restrict__ bias, void* __restrict__ Cout,
                 int M, int N, int K) {
  constexpr int BM = 128, BN = 128, BK = 64;
  __shared__ _Float16 As[BM * BK];        // [row][k]
  __shared__ _Float16 Bs[BN * BK];        // [n][k]

  const int tid = threadIdx.x;
  const int lane = tid & 31, wid = tid >> 5;
  const int bm0 = blockIdx.y * BM;
  const int bn0 = blockIdx.x * BN;
  const int wm0 = (wid & 1) * 64;
  const int wn0 = (wid >> 1) * 32;

  const int lr = tid >> 1;
  const int lc = (tid & 1) * 32;
  const _Float16* gA = A  + (size_t)(bm0 + lr) * K + lc;
  const _Float16* gB = Wt + (size_t)(bn0 + lr) * K + lc;

  v8h ra[4], rb[4];
#pragma unroll
  for (int i = 0; i < 4; ++i) {
    ra[i] = *(const v8h*)(gA + i * 8);
    rb[i] = *(const v8h*)(gB + i * 8);
  }

  v8f acc[4][2] = {};

  for (int k0 = 0; k0 < K; k0 += BK) {
#pragma unroll
    for (int i = 0; i < 4; ++i) {
      *(v8h*)&As[lr * BK + lc + i * 8] = ra[i];
      *(v8h*)&Bs[lr * BK + lc + i * 8] = rb[i];
    }
    __syncthreads();

    if (k0 + BK < K) {
      gA += BK; gB += BK;
#pragma unroll
      for (int i = 0; i < 4; ++i) {
        ra[i] = *(const v8h*)(gA + i * 8);
        rb[i] = *(const v8h*)(gB + i * 8);
      }
    }

#pragma unroll
    for (int kc = 0; kc < BK; kc += 32) {
      v16h af[4], bf[2];
#pragma unroll
      for (int mi = 0; mi < 4; ++mi) af[mi] = frag_a_rm(As, BK, wm0 + mi * 16, kc, lane);
#pragma unroll
      for (int ni = 0; ni < 2; ++ni) bf[ni] = frag_b_nm(Bs, BK, wn0 + ni * 16, kc, lane);
#pragma unroll
      for (int mi = 0; mi < 4; ++mi)
#pragma unroll
        for (int ni = 0; ni < 2; ++ni)
          acc[mi][ni] = WMMA_F16(af[mi], bf[ni], acc[mi][ni]);
    }
    __syncthreads();
  }

  const int lo = lane & 15, hi = lane >> 4;
#pragma unroll
  for (int ni = 0; ni < 2; ++ni) {
    int col = bn0 + wn0 + ni * 16 + lo;
    float bv = bias[col];
#pragma unroll
    for (int mi = 0; mi < 4; ++mi) {
#pragma unroll
      for (int j = 0; j < 8; ++j) {
        int row = bm0 + wm0 + mi * 16 + hi * 8 + j;
        float v = acc[mi][ni][j] + bv;
        if (OUT_F32)        ((float*)Cout)[(size_t)row * N + col] = v;
        else if (TRANS_OUT) ((_Float16*)Cout)[(size_t)col * M + row] = (_Float16)v;
        else                ((_Float16*)Cout)[(size_t)row * N + col] = (_Float16)v;
      }
    }
  }
}

// ---------------------------------------------------------------------------
// Flash attention with online softmax.
// Grid: (S/128 q-tiles, 32 heads, 2 batches), 256 threads = 8 waves, each
// wave owning a 16-row strip of the 128-row Q tile. K staged [key][d]; V is
// pre-transposed ([feature][token]) so the [d][key] tile is a pure b128 path.
// K/V tiles are register double-buffered; Q fragments hoisted out of the loop.
__global__ void __launch_bounds__(256)
attn_kernel(const _Float16* __restrict__ qh, const _Float16* __restrict__ kh,
            const _Float16* __restrict__ vt, _Float16* __restrict__ aoh) {
  constexpr int D = 64, BR = 128, BC = 64, S = 2048, HQ = 32, HKV = 8, G = 4;
  constexpr int QLD = HQ * D;             // 2048
  constexpr int KVLD = HKV * D;           // 512
  constexpr int M = 2 * S;                // B*S, row dim of vt

  __shared__ _Float16 Qs[BR * D];         // [row][d]   16 KB
  __shared__ _Float16 Ks[BC * D];         // [key][d]    8 KB
  __shared__ _Float16 Vs[D * BC];         // [d][key]    8 KB
  __shared__ _Float16 Ps[8][16 * BC];     // per-wave P 16 KB

  const int tid = threadIdx.x;
  const int lane = tid & 31, wid = tid >> 5;
  const int lo = lane & 15, hi = lane >> 4;
  const int qt = blockIdx.x, h = blockIdx.y, b = blockIdx.z;
  const int kvh = h / G;
  const int q0 = qt * BR;

  const _Float16* qptr = qh + (size_t)b * S * QLD + (size_t)h * D;
  const _Float16* kptr = kh + (size_t)b * S * KVLD + (size_t)kvh * D;
  const _Float16* vbase = vt + (size_t)(kvh * D) * M + (size_t)b * S;

  // Load Q tile (128x64 halves): thread moves 64B.
  {
    int r = tid >> 1;
    int c = (tid & 1) * 32;
    const _Float16* g = qptr + (size_t)(q0 + r) * QLD + c;
#pragma unroll
    for (int i = 0; i < 4; ++i)
      *(v8h*)&Qs[r * D + c + i * 8] = *(const v8h*)(g + i * 8);
  }
  __syncthreads();

  const int wr0 = wid * 16;
  // Q fragments are loop-invariant: hoist both K-chunks into registers.
  v16h aq[2];
#pragma unroll
  for (int kc = 0; kc < 2; ++kc) aq[kc] = frag_a_rm(Qs, D, wr0, kc * 32, lane);

  float mrow[8], lrow[8];
#pragma unroll
  for (int j = 0; j < 8; ++j) { mrow[j] = -3.0e38f; lrow[j] = 0.f; }
  v8f o[4] = {};

  _Float16* pw = &Ps[wid][0];
  const float scale = 0.125f;             // 1/sqrt(64)
  const int ktiles = 2 * qt + 2;          // causal truncation for 128-row tile

  // K/V tile staging coordinates: 64 rows x 64 halves, 16 halves/thread.
  const int r2 = tid >> 2;
  const int c2 = (tid & 3) * 16;

  // Stage first K/V tile into registers.
  v8h rk[2], rv[2];
  {
    const _Float16* gk = kptr + (size_t)r2 * KVLD + c2;
    const _Float16* gv = vbase + (size_t)r2 * M + c2;
#pragma unroll
    for (int i = 0; i < 2; ++i) {
      rk[i] = *(const v8h*)(gk + i * 8);
      rv[i] = *(const v8h*)(gv + i * 8);
    }
  }

  for (int kt = 0; kt < ktiles; ++kt) {
    const int kc0 = kt * BC;
    // Commit staged K/V tile to LDS.
#pragma unroll
    for (int i = 0; i < 2; ++i) {
      *(v8h*)&Ks[r2 * D + c2 + i * 8] = rk[i];
      *(v8h*)&Vs[r2 * BC + c2 + i * 8] = rv[i];
    }
    __syncthreads();

    // Stage next tile while WMMAs run; prefetch the tile after next.
    if (kt + 1 < ktiles) {
      const _Float16* gk = kptr + (size_t)(kc0 + BC + r2) * KVLD + c2;
      const _Float16* gv = vbase + (size_t)r2 * M + kc0 + BC + c2;
#pragma unroll
      for (int i = 0; i < 2; ++i) {
        rk[i] = *(const v8h*)(gk + i * 8);
        rv[i] = *(const v8h*)(gv + i * 8);
      }
      if (kt + 2 < ktiles) {
        __builtin_prefetch(kptr + (size_t)(kc0 + 2 * BC + r2) * KVLD + c2, 0, 3);
        __builtin_prefetch(vbase + (size_t)r2 * M + kc0 + 2 * BC + c2, 0, 3);
      }
    }

    // Wave-uniform causal early-out: this wave's strip needs this key tile
    // only if kc0 <= max row of the strip.
    if (kc0 <= q0 + wr0 + 15) {
      // S strip = Q_strip(16xD) * K^T: 4 col-tiles x 2 K-chunks.
      v8f st[4] = {};
#pragma unroll
      for (int kc = 0; kc < 2; ++kc) {
#pragma unroll
        for (int nt = 0; nt < 4; ++nt) {
          v16h bk = frag_b_nm(Ks, D, nt * 16, kc * 32, lane);
          st[nt] = WMMA_F16(aq[kc], bk, st[nt]);
        }
      }

      // Scale; mask only when the tile overlaps this strip's diagonal.
      if (kc0 + BC - 1 > q0 + wr0) {
#pragma unroll
        for (int nt = 0; nt < 4; ++nt) {
          int kc = kc0 + nt * 16 + lo;
#pragma unroll
          for (int j = 0; j < 8; ++j) {
            int qr = q0 + wr0 + hi * 8 + j;
            float s = st[nt][j] * scale;
            st[nt][j] = (kc > qr) ? -3.0e38f : s;
          }
        }
      } else {
#pragma unroll
        for (int nt = 0; nt < 4; ++nt)
#pragma unroll
          for (int j = 0; j < 8; ++j)
            st[nt][j] *= scale;
      }

      // Row-wise max across this key tile (16-lane half-group reduction).
      float tmax[8];
#pragma unroll
      for (int j = 0; j < 8; ++j)
        tmax[j] = fmaxf(fmaxf(st[0][j], st[1][j]), fmaxf(st[2][j], st[3][j]));
#pragma unroll
      for (int off = 1; off < 16; off <<= 1)
#pragma unroll
        for (int j = 0; j < 8; ++j)
          tmax[j] = fmaxf(tmax[j], __shfl_xor(tmax[j], off, 32));

      float mnew[8], corr[8];
#pragma unroll
      for (int j = 0; j < 8; ++j) {
        mnew[j] = fmaxf(mrow[j], tmax[j]);
        corr[j] = __expf(mrow[j] - mnew[j]);
        mrow[j] = mnew[j];
      }

      // P = exp(S - mnew); row sums.
      float tsum[8];
#pragma unroll
      for (int j = 0; j < 8; ++j) tsum[j] = 0.f;
#pragma unroll
      for (int nt = 0; nt < 4; ++nt)
#pragma unroll
        for (int j = 0; j < 8; ++j) {
          float p = __expf(st[nt][j] - mnew[j]);
          st[nt][j] = p;
          tsum[j] += p;
        }
#pragma unroll
      for (int off = 1; off < 16; off <<= 1)
#pragma unroll
        for (int j = 0; j < 8; ++j)
          tsum[j] += __shfl_xor(tsum[j], off, 32);
#pragma unroll
      for (int j = 0; j < 8; ++j) lrow[j] = lrow[j] * corr[j] + tsum[j];

      // Rescale O accumulators (per-row factors are lane-local in C layout).
#pragma unroll
      for (int nt = 0; nt < 4; ++nt)
#pragma unroll
        for (int j = 0; j < 8; ++j) o[nt][j] *= corr[j];

      // Stage P (f16) into this wave's LDS strip (C-layout -> row-major).
#pragma unroll
      for (int nt = 0; nt < 4; ++nt)
#pragma unroll
        for (int j = 0; j < 8; ++j)
          pw[(size_t)(hi * 8 + j) * BC + nt * 16 + lo] = (_Float16)st[nt][j];
      // Wave-local LDS ops are in-order: safe to read back without a barrier.

      // O += P(16x64) * V(64xD).
#pragma unroll
      for (int kc = 0; kc < 2; ++kc) {
        v16h ap = frag_a_rm(pw, BC, 0, kc * 32, lane);
#pragma unroll
        for (int nt = 0; nt < 4; ++nt) {
          v16h bv = frag_b_nm(Vs, BC, nt * 16, kc * 32, lane);
          o[nt] = WMMA_F16(ap, bv, o[nt]);
        }
      }
    }
    __syncthreads();                       // before Ks/Vs are overwritten
  }

  // Epilogue: O /= l, store f16 [B,S,HQ*D].
#pragma unroll
  for (int j = 0; j < 8; ++j) lrow[j] = 1.0f / lrow[j];
#pragma unroll
  for (int nt = 0; nt < 4; ++nt)
#pragma unroll
    for (int j = 0; j < 8; ++j) {
      int row = q0 + wr0 + hi * 8 + j;
      int col = h * D + nt * 16 + lo;
      aoh[(size_t)(b * S + row) * QLD + col] = (_Float16)(o[nt][j] * lrow[j]);
    }
}

// ---------------------------------------------------------------------------
extern "C" void kernel_launch(void* const* d_in, const int* in_sizes, int n_in,
                              void* d_out, int out_size, void* d_ws, size_t ws_size,
                              hipStream_t stream) {
  (void)in_sizes; (void)n_in; (void)out_size; (void)ws_size;
  constexpr int B = 2, S = 2048, HID = 2048, HQ = 32, HKV = 8, D = 64;
  const int M = B * S;                     // 4096
  const int NKV = HKV * D;                 // 512

  const float* x  = (const float*)d_in[0];
  // d_in[1] = causal mask (recomputed analytically in-kernel)
  const float* Wq = (const float*)d_in[2];
  const float* bq = (const float*)d_in[3];
  const float* Wk = (const float*)d_in[4];
  const float* bk = (const float*)d_in[5];
  const float* Wv = (const float*)d_in[6];
  const float* bv = (const float*)d_in[7];
  const float* Wo = (const float*)d_in[8];
  const float* bo = (const float*)d_in[9];
  float* out = (float*)d_out;

  // f16 staging in workspace (~80 MB).
  _Float16* ws = (_Float16*)d_ws;
  size_t off = 0;
  _Float16* xh  = ws + off; off += (size_t)M * HID;     // x (f16)
  _Float16* wqT = ws + off; off += (size_t)HID * HID;   // Wq^T [N][K]
  _Float16* wkT = ws + off; off += (size_t)NKV * HID;   // Wk^T
  _Float16* wvT = ws + off; off += (size_t)NKV * HID;   // Wv^T
  _Float16* woT = ws + off; off += (size_t)HID * HID;   // Wo^T
  _Float16* qh  = ws + off; off += (size_t)M * HID;     // Q [token][feat]
  _Float16* kh  = ws + off; off += (size_t)M * NKV;     // K [token][feat]
  _Float16* vt  = ws + off; off += (size_t)NKV * M;     // V^T [feat][token]
  _Float16* aoh = ws + off; off += (size_t)M * HID;     // attention out

  {
    size_t n = (size_t)M * HID;
    int blocks = (int)((n / 4 + 255) / 256); if (blocks > 4096) blocks = 4096;
    f32_to_f16_kernel<<<blocks, 256, 0, stream>>>(x, xh, n);
  }
  transpose_cvt_kernel<<<dim3(HID / 64, HID / 64), 256, 0, stream>>>(Wq, wqT, HID, HID);
  transpose_cvt_kernel<<<dim3(HID / 64, NKV / 64), 256, 0, stream>>>(Wk, wkT, HID, NKV);
  transpose_cvt_kernel<<<dim3(HID / 64, NKV / 64), 256, 0, stream>>>(Wv, wvT, HID, NKV);
  transpose_cvt_kernel<<<dim3(HID / 64, HID / 64), 256, 0, stream>>>(Wo, woT, HID, HID);

  // QKV projections (V written transposed for the attention PV stage).
  gemm_bias_kernel<false, false><<<dim3(HID / 128, M / 128), 256, 0, stream>>>(xh, wqT, bq, qh, M, HID, HID);
  gemm_bias_kernel<false, false><<<dim3(NKV / 128, M / 128), 256, 0, stream>>>(xh, wkT, bk, kh, M, NKV, HID);
  gemm_bias_kernel<false, true ><<<dim3(NKV / 128, M / 128), 256, 0, stream>>>(xh, wvT, bv, vt, M, NKV, HID);

  // Flash attention.
  attn_kernel<<<dim3(S / 128, HQ, B), 256, 0, stream>>>(qh, kh, vt, aoh);

  // Output projection (f32 result + bias).
  gemm_bias_kernel<true, false><<<dim3(HID / 128, M / 128), 256, 0, stream>>>(aoh, woT, bo, out, M, HID, HID);
}
